// NPU_FusedMLAPreprocess_10488310137174
// MI455X (gfx1250) — 16-bit and 8-bit matrix operand order repaired and checked by reading, compiled, not run on hardware
//
#include <hip/hip_runtime.h>
#include <stdint.h>

typedef __attribute__((ext_vector_type(8)))  int          v8i;
typedef __attribute__((ext_vector_type(4)))  int          v4i;
typedef __attribute__((ext_vector_type(4)))  unsigned int v4u;
typedef __attribute__((ext_vector_type(16))) _Float16     v16h;
typedef __attribute__((ext_vector_type(8)))  float        v8f;
typedef __attribute__((ext_vector_type(2)))  float        v2f;

#define T_TOK 4096
#define HID   7168
#define QL    1536
#define KVL   512
#define ROPE  64
#define NOPE  128
#define NHEAD 16
#define N0    2112
#define NQ    3072
#define SLOTS 16384
#define EPS   1e-6f

#if __has_builtin(__builtin_amdgcn_tensor_load_to_lds) && \
    __has_builtin(__builtin_amdgcn_s_wait_tensorcnt)
#define HAS_TDM 1
#else
#define HAS_TDM 0
#endif

__global__ __launch_bounds__(256) void k_rmsnorm_quant0(
    const float* __restrict__ x, const float* __restrict__ g,
    const float* __restrict__ b, const float* __restrict__ qs,
    const int* __restrict__ qo, int8_t* __restrict__ out)
{
  __shared__ float red[256];
  const int t = blockIdx.x;
  const float* row = x + (size_t)t * HID;
  float s = 0.f;
  for (int i = threadIdx.x; i < HID; i += 256) { float v = row[i]; s += v * v; }
  red[threadIdx.x] = s; __syncthreads();
  for (int off = 128; off > 0; off >>= 1) {
    if (threadIdx.x < off) red[threadIdx.x] += red[threadIdx.x + off];
    __syncthreads();
  }
  const float rs    = rsqrtf(red[0] / (float)HID + EPS);
  const float scale = qs[0];
  const float offq  = (float)qo[0];
  int8_t* orow = out + (size_t)t * HID;
  for (int i = threadIdx.x; i < HID; i += 256) {
    float v = row[i] * rs * g[i] + b[i];
    float q = rintf(v / scale + offq);
    q = fminf(fmaxf(q, -128.f), 127.f);
    orow[i] = (int8_t)q;
  }
}

#if HAS_TDM
__device__ __forceinline__ void tdm_load_tile(unsigned int lds_addr,
                                              const int8_t* gptr,
                                              int rows, int ldk)
{
  const unsigned long long ga = (unsigned long long)(uintptr_t)gptr;
  v4u g0;
  g0[0] = 1u;
  g0[1] = lds_addr;
  g0[2] = (unsigned int)ga;
  g0[3] = (unsigned int)((ga >> 32) & 0x1ffffffu)
          | (2u << 30);
  v8i g1;
  g1[0] = 0;
  g1[1] = (int)(64u << 16);
  g1[2] = (int)(((unsigned)rows & 0xffffu) << 16);
  g1[3] = (int)(64u << 16);
  g1[4] = rows & 0xffff;
  g1[5] = ldk;
  g1[6] = 0;
  g1[7] = 0;
  v4i z4 = {0, 0, 0, 0};
#if defined(__clang_major__) && (__clang_major__ >= 23)
  v8i z8 = (v8i){};
  __builtin_amdgcn_tensor_load_to_lds(g0, g1, z4, z4, z8, 0);
#else
  __builtin_amdgcn_tensor_load_to_lds(g0, g1, z4, z4, 0);
#endif
}
#else
__device__ __forceinline__ void coop_load_tile(int8_t* dst, const int8_t* gsrc,
                                               int rows, int ldk, int tid)
{
  const int nvec = rows * 4;
  for (int v = tid; v < nvec; v += 256) {
    const int r = v >> 2, c = (v & 3) << 4;
    *(v4i*)(dst + r * 64 + c) = *(const v4i*)(gsrc + (size_t)r * ldk + c);
  }
}
#endif

__global__ __launch_bounds__(256) void k_gemm_i8(
    const int8_t* __restrict__ A, const int8_t* __restrict__ W,
    const int* __restrict__ bias, const float* __restrict__ descale,
    float* __restrict__ out, int Ncols, int K)
{
  __shared__ __align__(16) int8_t As[2][128 * 64];
  __shared__ __align__(16) int8_t Bs[2][64 * 64];

  const int tid  = threadIdx.x;
  const int lane = tid & 31;
  const int w    = tid >> 5;
  const int wm   = w & 3, wn = w >> 2;
  const int r15  = lane & 15;
  const int g    = lane >> 4;

  const int mblk = blockIdx.y * 128;
  const int nblk = blockIdx.x * 64;
  const int8_t* Ab = A + (size_t)mblk * K;
  const int8_t* Wb = W + (size_t)nblk * K;

#if HAS_TDM
  if (w == 0) {
    tdm_load_tile((unsigned int)(uintptr_t)&As[0][0], Ab, 128, K);
    tdm_load_tile((unsigned int)(uintptr_t)&Bs[0][0], Wb, 64, K);
    __builtin_amdgcn_s_wait_tensorcnt(0);
  }
#else
  coop_load_tile(&As[0][0], Ab, 128, K, tid);
  coop_load_tile(&Bs[0][0], Wb, 64, K, tid);
#endif
  __syncthreads();

  v8i c[2][2];
  c[0][0] = (v8i){}; c[0][1] = (v8i){}; c[1][0] = (v8i){}; c[1][1] = (v8i){};

  int buf = 0;
  for (int kk = 0; kk < K; kk += 64) {
    const int nxt = buf ^ 1;
    const bool more = (kk + 64) < K;
    if (more) {
#if HAS_TDM
      if (w == 0) {
        tdm_load_tile((unsigned int)(uintptr_t)&As[nxt][0], Ab + kk + 64, 128, K);
        tdm_load_tile((unsigned int)(uintptr_t)&Bs[nxt][0], Wb + kk + 64, 64, K);
      }
#else
      coop_load_tile(&As[nxt][0], Ab + kk + 64, 128, K, tid);
      coop_load_tile(&Bs[nxt][0], Wb + kk + 64, 64, K, tid);
#endif
    }

    const int8_t* Ap0 = &As[buf][(wm * 32 + r15) * 64];
    const int8_t* Ap1 = &As[buf][(wm * 32 + 16 + r15) * 64];
    const int8_t* Bp0 = &Bs[buf][(wn * 32 + r15) * 64];
    const int8_t* Bp1 = &Bs[buf][(wn * 32 + 16 + r15) * 64];
    v8i a0, a1, b0, b1;
#pragma unroll
    for (int i = 0; i < 8; ++i) {
      const int ka = ((i >> 1) << 4) + (g << 3) + ((i & 1) << 2);
      a0[i] = *(const int*)(Ap0 + ka);
      a1[i] = *(const int*)(Ap1 + ka);
      const int kb = ((i >> 1) << 4) + (g << 3) + ((i & 1) << 2);
      b0[i] = *(const int*)(Bp0 + kb);
      b1[i] = *(const int*)(Bp1 + kb);
    }

    c[0][0] = __builtin_amdgcn_wmma_i32_16x16x64_iu8(true, a0, true, b0, c[0][0], false, false);
    c[0][1] = __builtin_amdgcn_wmma_i32_16x16x64_iu8(true, a0, true, b1, c[0][1], false, false);
    c[1][0] = __builtin_amdgcn_wmma_i32_16x16x64_iu8(true, a1, true, b0, c[1][0], false, false);
    c[1][1] = __builtin_amdgcn_wmma_i32_16x16x64_iu8(true, a1, true, b1, c[1][1], false, false);

#if HAS_TDM
    if (more && w == 0) __builtin_amdgcn_s_wait_tensorcnt(0);
#endif
    __syncthreads();
    buf = nxt;
  }

  const int halfsel = g << 3;
#pragma unroll
  for (int mi = 0; mi < 2; ++mi) {
#pragma unroll
    for (int ni = 0; ni < 2; ++ni) {
      const int   col = nblk + wn * 32 + ni * 16 + r15;
      const float ds  = descale[col];
      const int   bs  = bias[col];
#pragma unroll
      for (int r = 0; r < 8; ++r) {
        const int row = mblk + wm * 32 + mi * 16 + r + halfsel;
        out[(size_t)row * Ncols + col] = (float)(c[mi][ni][r] + bs) * ds;
      }
    }
  }
}

__global__ __launch_bounds__(256) void k_split(
    const float* __restrict__ out0, const float* __restrict__ gamma2,
    const float* __restrict__ gamma1, const float* __restrict__ beta1,
    const float* __restrict__ cosb, const float* __restrict__ sinb,
    const float* __restrict__ qs, const int* __restrict__ qo,
    const int* __restrict__ slot_mapping,
    float* __restrict__ kv_out, float* __restrict__ kvrope_out,
    int8_t* __restrict__ q8)
{
  __shared__ float redk[256];
  __shared__ float redq[256];
  const int t = blockIdx.x;
  const float* row = out0 + (size_t)t * N0;

  float sk = 0.f, sq = 0.f;
  for (int i = threadIdx.x; i < KVL; i += 256) { float v = row[i];              sk += v * v; }
  for (int i = threadIdx.x; i < QL;  i += 256) { float v = row[KVL + ROPE + i]; sq += v * v; }
  redk[threadIdx.x] = sk; redq[threadIdx.x] = sq; __syncthreads();
  for (int off = 128; off > 0; off >>= 1) {
    if (threadIdx.x < off) {
      redk[threadIdx.x] += redk[threadIdx.x + off];
      redq[threadIdx.x] += redq[threadIdx.x + off];
    }
    __syncthreads();
  }
  const float rsk  = rsqrtf(redk[0] / (float)KVL + EPS);
  const float rsq  = rsqrtf(redq[0] / (float)QL  + EPS);
  const int   slot = slot_mapping[t];

  for (int i = threadIdx.x; i < KVL; i += 256)
    kv_out[(size_t)slot * KVL + i] = row[i] * rsk * gamma2[i];

  for (int i = threadIdx.x; i < ROPE; i += 256) {
    const float x   = row[KVL + i];
    const float rot = (i < ROPE / 2) ? -row[KVL + i + ROPE / 2]
                                     :  row[KVL + i - ROPE / 2];
    kvrope_out[(size_t)slot * ROPE + i] =
        x * cosb[t * ROPE + i] + rot * sinb[t * ROPE + i];
  }

  const float scale = qs[0];
  const float offq  = (float)qo[0];
  for (int i = threadIdx.x; i < QL; i += 256) {
    float v = row[KVL + ROPE + i] * rsq * gamma1[i] + beta1[i];
    float q = rintf(v / scale + offq);
    q8[(size_t)t * QL + i] = (int8_t)fminf(fmaxf(q, -128.f), 127.f);
  }
}

__global__ __launch_bounds__(256) void k_qrope(
    const float* __restrict__ q, const float* __restrict__ cosb,
    const float* __restrict__ sinb, float* __restrict__ out)
{
  const int t = blockIdx.x;
  for (int idx = threadIdx.x; idx < NHEAD * ROPE; idx += 256) {
    const int h = idx >> 6;
    const int i = idx & 63;
    const float* base = q + (size_t)t * NQ + h * (NOPE + ROPE) + NOPE;
    const float x   = base[i];
    const float rot = (i < 32) ? -base[i + 32] : base[i - 32];
    out[((size_t)t * NHEAD + h) * ROPE + i] =
        x * cosb[t * ROPE + i] + rot * sinb[t * ROPE + i];
  }
}

__global__ __launch_bounds__(256) void k_einsum(
    const float* __restrict__ q, const float* __restrict__ wkc,
    float* __restrict__ out)
{
  const int lane  = threadIdx.x & 31;
  const int w     = threadIdx.x >> 5;
  const int h     = blockIdx.z;
  const int mbase = blockIdx.y * 16;
  const int nbase = blockIdx.x * 256 + w * 32;
  const int r15   = lane & 15;
  const int g     = lane >> 4;

  v8f c0 = {}, c1 = {};
  const float* arow = q + (size_t)(mbase + r15) * NQ + h * (NOPE + ROPE);
  const float* wh   = wkc + (size_t)h * NOPE * KVL;

  for (int kk = 0; kk < NOPE; kk += 32) {
    v16h a;
#pragma unroll
    for (int j = 0; j < 8; ++j) {
      const int k = kk + (g << 4) + 2 * j;
      const v2f f = *(const v2f*)(arow + k);
      a[2 * j]     = (_Float16)f.x;
      a[2 * j + 1] = (_Float16)f.y;
    }
    v16h b0, b1;
#pragma unroll
    for (int j = 0; j < 8; ++j) {
      const int k = kk + (g << 4) + 2 * j;
      b0[2 * j]     = (_Float16)wh[(size_t)k       * KVL + nbase + r15];
      b0[2 * j + 1] = (_Float16)wh[(size_t)(k + 1) * KVL + nbase + r15];
      b1[2 * j]     = (_Float16)wh[(size_t)k       * KVL + nbase + 16 + r15];
      b1[2 * j + 1] = (_Float16)wh[(size_t)(k + 1) * KVL + nbase + 16 + r15];
    }
    c0 = __builtin_amdgcn_wmma_f32_16x16x32_f16(false, a, false, b0, (short)0, c0, false, false);
    c1 = __builtin_amdgcn_wmma_f32_16x16x32_f16(false, a, false, b1, (short)0, c1, false, false);
  }

  const int halfsel = g << 3;
#pragma unroll
  for (int r = 0; r < 8; ++r) {
    const int row = mbase + r + halfsel;
    out[((size_t)row * NHEAD + h) * KVL + nbase + r15]      = c0[r];
    out[((size_t)row * NHEAD + h) * KVL + nbase + 16 + r15] = c1[r];
  }
}

extern "C" void kernel_launch(void* const* d_in, const int* in_sizes, int n_in,
                              void* d_out, int out_size, void* d_ws, size_t ws_size,
                              hipStream_t stream)
{
  (void)in_sizes; (void)n_in; (void)out_size; (void)ws_size;

  const float*  hidden   = (const float*)  d_in[0];
  const float*  gamma0   = (const float*)  d_in[1];
  const float*  beta0    = (const float*)  d_in[2];
  const int8_t* wdqkv    = (const int8_t*) d_in[3];
  const float*  descale0 = (const float*)  d_in[4];
  const int*    bias0    = (const int*)    d_in[5];
  const float*  qscale0  = (const float*)  d_in[6];
  const int*    qoff0    = (const int*)    d_in[7];
  const float*  gamma1   = (const float*)  d_in[8];
  const float*  beta1    = (const float*)  d_in[9];
  const int8_t* wuq      = (const int8_t*) d_in[10];
  const float*  descale1 = (const float*)  d_in[11];
  const int*    bias1    = (const int*)    d_in[12];
  const float*  qscale1  = (const float*)  d_in[13];
  const int*    qoff1    = (const int*)    d_in[14];
  const float*  gamma2   = (const float*)  d_in[15];
  const float*  cosb     = (const float*)  d_in[16];
  const float*  sinb     = (const float*)  d_in[17];
  const float*  w_kc     = (const float*)  d_in[18];
  const float*  kv_in    = (const float*)  d_in[19];
  const float*  kvr_in   = (const float*)  d_in[20];
  const int*    slots    = (const int*)    d_in[21];

  float* outf = (float*)d_out;
  const size_t off_qrope  = 0;
  const size_t off_kvrope = (size_t)T_TOK * NHEAD * ROPE;
  const size_t off_qnope  = off_kvrope + (size_t)SLOTS * ROPE;
  const size_t off_kv     = off_qnope  + (size_t)T_TOK * NHEAD * KVL;

  char* ws = (char*)d_ws;
  size_t cur = 0;
  auto alloc = [&](size_t bytes) {
    char* p = ws + cur;
    cur += (bytes + 255) & ~(size_t)255;
    return p;
  };
  int8_t* x8   = (int8_t*)alloc((size_t)T_TOK * HID);
  float*  out0 = (float*) alloc((size_t)T_TOK * N0 * 4);
  int8_t* q8   = (int8_t*)alloc((size_t)T_TOK * QL);
  float*  qws  = (float*) alloc((size_t)T_TOK * NQ * 4);

  hipMemcpyAsync(outf + off_kv,     kv_in,  (size_t)SLOTS * KVL  * 4,
                 hipMemcpyDeviceToDevice, stream);
  hipMemcpyAsync(outf + off_kvrope, kvr_in, (size_t)SLOTS * ROPE * 4,
                 hipMemcpyDeviceToDevice, stream);

  k_rmsnorm_quant0<<<T_TOK, 256, 0, stream>>>(hidden, gamma0, beta0,
                                              qscale0, qoff0, x8);
  k_gemm_i8<<<dim3(N0 / 64, T_TOK / 128), 256, 0, stream>>>(
      x8, wdqkv, bias0, descale0, out0, N0, HID);
  k_split<<<T_TOK, 256, 0, stream>>>(out0, gamma2, gamma1, beta1, cosb, sinb,
                                     qscale1, qoff1, slots,
                                     outf + off_kv, outf + off_kvrope, q8);
  k_gemm_i8<<<dim3(NQ / 64, T_TOK / 128), 256, 0, stream>>>(
      q8, wuq, bias1, descale1, qws, NQ, QL);
  k_qrope<<<T_TOK, 256, 0, stream>>>(qws, cosb, sinb, outf + off_qrope);
  k_einsum<<<dim3(KVL / 256, T_TOK / 16, NHEAD), 256, 0, stream>>>(
      qws, w_kc, outf + off_qnope);
}
